// FeedForwardQuantum_65481071406694
// MI455X (gfx1250) — matrix operand order repaired and checked by reading, not compiled, not run on hardware
//
#include <hip/hip_runtime.h>
#include <hip/hip_bf16.h>

typedef __attribute__((ext_vector_type(16))) _Float16     v16h;
typedef __attribute__((ext_vector_type(8)))  float        v8f;
typedef __attribute__((ext_vector_type(4)))  unsigned int u32x4;

#define BQ       8
#define E_DIM    512
#define FFN_DIM  2048
#define BS_ROWS  8192          // B*S = 4*2048
#define M_TILE   64            // rows per workgroup
#define KC       64            // FFN chunk staged in LDS
#define HPAD     72            // 64 + 8 halfs padding (16B) -> conflict-free ds_load_b128

union HalfFrag { v16h h; u32x4 q[2]; };

// ---------------------------------------------------------------------------
// Kernel 1: W2 (FFN x E, f32 row-major) -> W2T (E x FFN, f16) in workspace.
// Makes GEMM2 B-fragments contiguous-32B-per-lane loads.
// ---------------------------------------------------------------------------
__global__ void w2_transpose_f16(const float* __restrict__ W2,
                                 _Float16* __restrict__ w2t) {
  int i = blockIdx.x * blockDim.x + threadIdx.x;   // 0 .. E*FFN-1
  int e = i >> 11;              // / FFN_DIM
  int f = i & (FFN_DIM - 1);
  w2t[i] = (_Float16)W2[f * E_DIM + e];
}

// ---------------------------------------------------------------------------
// Kernel 2: fused  z = cos(theta)*cos(x)  ->  h = relu(z@W1+b1) (VALU, K=8)
//           ->  out = h@W2 + b2  (v_wmma_f32_16x16x32_f16, K=2048)
// 256 threads (8 wave32). Workgroup tile 64x512; each wave owns 64x64
// (4 M-tiles x 4 N-tiles) so every B fragment feeds 4 WMMAs.
// ---------------------------------------------------------------------------
__global__ __launch_bounds__(256)
void ffq_fused(const float* __restrict__ x,
               const float* __restrict__ theta,
               const float* __restrict__ W1,
               const float* __restrict__ b1,
               const _Float16* __restrict__ w2t,
               const float* __restrict__ b2,
               float* __restrict__ out) {
  __shared__ __align__(16) float     zc[M_TILE][BQ];
  __shared__ __align__(16) _Float16  hbuf[2][M_TILE][HPAD];

  const int tid     = threadIdx.x;
  const int rowbase = blockIdx.x * M_TILE;

  // ---- phase 0: z tile = cos(theta[q]) * cos(x[row][q]) ----
  for (int i = tid; i < M_TILE * BQ; i += 256) {
    int r = i >> 3, q = i & 7;
    zc[r][q] = __cosf(theta[q]) * __cosf(x[(rowbase + r) * E_DIM + q]);
  }
  __syncthreads();   // zc is read cross-thread by the producer below

  // wave tiling: each of 8 waves owns all 64 rows x a 64-col slice
  const int wave  = tid >> 5;
  const int lane  = tid & 31;
  const int nbase = wave * 64;
  const int lmod  = lane & 15;
  const int lhi   = (lane >> 4) & 1;

  // producer mapping: 256 threads cover 64 cols x 4 row-groups of 16
  const int floc = tid & 63;
  const int rgrp = (tid >> 6) * 16;

  v8f acc[4][4];
  {
    v8f zero = {};
#pragma unroll
    for (int mt = 0; mt < 4; ++mt)
#pragma unroll
      for (int nt = 0; nt < 4; ++nt) acc[mt][nt] = zero;
  }

  // ---- producer: h[64][KC] = relu(z @ W1 + b1), f16 into LDS buffer ----
  auto produce = [&](_Float16 (*hb)[HPAD], int kc) {
    const int f = kc + floc;
    float w1c[BQ];
#pragma unroll
    for (int q = 0; q < BQ; ++q) w1c[q] = W1[q * FFN_DIM + f];
    const float bb = b1[f];
#pragma unroll
    for (int r = 0; r < 16; ++r) {
      float s = bb;
#pragma unroll
      for (int q = 0; q < BQ; ++q) s += zc[rgrp + r][q] * w1c[q];
      s = s > 0.0f ? s : 0.0f;
      hb[rgrp + r][floc] = (_Float16)s;
    }
  };

  // ---- consumer: 32 WMMAs over one K chunk ----
  auto consume = [&](const _Float16 (*hb)[HPAD], int kc) {
#pragma unroll
    for (int ks = 0; ks < KC; ks += 32) {
      // B fragments first (long-latency global/L2 loads).
      // B (32x16): lane = column n, 16 contiguous K halfs; lanes 16-31
      // hold K+16..K+31 -> 32B contiguous from W2T row n.
      HalfFrag bf[4];
      const _Float16* wbase =
          w2t + (nbase + lmod) * FFN_DIM + kc + ks + (lhi ? 16 : 0);
#pragma unroll
      for (int nt = 0; nt < 4; ++nt) {
        const u32x4* pb = (const u32x4*)(wbase + nt * 16 * FFN_DIM);
        bf[nt].q[0] = pb[0];
        bf[nt].q[1] = pb[1];
      }
      // A fragments (contiguous K: lanes16-31 hold K+16..K+31)
      HalfFrag a[4];
      const int kb = ks + (lhi ? 16 : 0);
#pragma unroll
      for (int mt = 0; mt < 4; ++mt) {
        const u32x4* pa = (const u32x4*)&hb[mt * 16 + lmod][kb];
        a[mt].q[0] = pa[0];      // halfs kb   .. kb+7
        a[mt].q[1] = pa[1];      // halfs kb+8 .. kb+15
      }
      // 16 WMMAs; each bf[nt] feeds 4, each a[mt] feeds 4
#pragma unroll
      for (int nt = 0; nt < 4; ++nt)
#pragma unroll
        for (int mt = 0; mt < 4; ++mt)
          acc[mt][nt] = __builtin_amdgcn_wmma_f32_16x16x32_f16(
              false, a[mt].h, false, bf[nt].h, (short)0, acc[mt][nt],
              false, false);
    }
  };

  // ---- main loop: double-buffered, one barrier per K chunk ----
  // (FFN_DIM / KC = 32 chunks, manually unrolled x2 so buffer indices are
  //  compile-time constants -> LDS alias analysis can overlap phases)
  produce(hbuf[0], 0);
  for (int kc = 0; kc < FFN_DIM; kc += 2 * KC) {
    __syncthreads();
    __builtin_prefetch(w2t + (nbase + lmod) * FFN_DIM + kc + KC, 0, 3);
    produce(hbuf[1], kc + KC);          // fill other buffer
    consume(hbuf[0], kc);               // WMMA current buffer
    __syncthreads();
    if (kc + 2 * KC < FFN_DIM) {
      __builtin_prefetch(w2t + (nbase + lmod) * FFN_DIM + kc + 2 * KC, 0, 3);
      produce(hbuf[0], kc + 2 * KC);
    }
    consume(hbuf[1], kc + KC);
  }

  // ---- epilogue: out = acc + b2 (C/D layout: VGPR j -> M=j / j+8) ----
#pragma unroll
  for (int nt = 0; nt < 4; ++nt) {
    const int n  = nbase + nt * 16 + lmod;
    const float bv = b2[n];
#pragma unroll
    for (int mt = 0; mt < 4; ++mt) {
      const int rb = rowbase + mt * 16 + (lhi ? 8 : 0);
#pragma unroll
      for (int j = 0; j < 8; ++j)
        out[(rb + j) * E_DIM + n] = acc[mt][nt][j] + bv;
    }
  }
}

// ---------------------------------------------------------------------------
extern "C" void kernel_launch(void* const* d_in, const int* in_sizes, int n_in,
                              void* d_out, int out_size, void* d_ws, size_t ws_size,
                              hipStream_t stream) {
  (void)in_sizes; (void)n_in; (void)out_size; (void)ws_size;
  const float* x     = (const float*)d_in[0];   // (4,2048,512) f32
  const float* theta = (const float*)d_in[1];   // (8,)         f32
  const float* W1    = (const float*)d_in[2];   // (8,2048)     f32
  const float* b1    = (const float*)d_in[3];   // (2048,)      f32
  const float* W2    = (const float*)d_in[4];   // (2048,512)   f32
  const float* b2    = (const float*)d_in[5];   // (512,)       f32
  float* out = (float*)d_out;                   // (4,2048,512) f32

  _Float16* w2t = (_Float16*)d_ws;              // 2 MB scratch

  w2_transpose_f16<<<(E_DIM * FFN_DIM) / 256, 256, 0, stream>>>(W2, w2t);
  ffq_fused<<<BS_ROWS / M_TILE, 256, 0, stream>>>(x, theta, W1, b1, w2t, b2, out);
}
